// ModConv2D_4629974745358
// MI455X (gfx1250) — compile-verified, matrix operand order checked by reading, not run on hardware
//
#include <hip/hip_runtime.h>
#include <hip/hip_bf16.h>

typedef __attribute__((ext_vector_type(8)))  _Float16 v8h;
typedef __attribute__((ext_vector_type(16))) _Float16 v16h;
typedef __attribute__((ext_vector_type(8)))  float    v8f;

#define BATCH 16
#define HH 64
#define WW 64
#define CINC 256
#define COUTC 256
#define KTOT 2304   // 9 * 256

// ---------------- kernel 1: x fp32 -> fp16 (same NHWC layout) ----------------
__global__ __launch_bounds__(256) void cvt_x_kernel(const float* __restrict__ x,
                                                    _Float16* __restrict__ x16) {
  size_t i = ((size_t)blockIdx.x * 256 + threadIdx.x) * 8;
  float4 a = *(const float4*)(x + i);
  float4 b = *(const float4*)(x + i + 4);
  v8h o;
  o[0] = (_Float16)a.x; o[1] = (_Float16)a.y; o[2] = (_Float16)a.z; o[3] = (_Float16)a.w;
  o[4] = (_Float16)b.x; o[5] = (_Float16)b.y; o[6] = (_Float16)b.z; o[7] = (_Float16)b.w;
  *(v8h*)(x16 + i) = o;
}

// ------------- kernel 2: demodulation scale per (b, co) -------------
__global__ __launch_bounds__(256) void sumsq_kernel(const float* __restrict__ kern,
                                                    const float* __restrict__ style,
                                                    float* __restrict__ scale) {
  const int b  = blockIdx.x;
  const int co = threadIdx.x;
  float acc = 0.0f;
  for (int k = 0; k < KTOT; ++k) {
    float s = style[b * CINC + (k & 255)] + 1.0f;   // uniform -> scalar load
    float v = kern[(size_t)k * COUTC + co] * s;
    acc += v * v;
  }
  scale[b * COUTC + co] = rsqrtf(acc + 1e-8f);
}

// ------------- kernel 3: modulated + demodulated weights, transposed to [b][co][k] f16 -------------
__global__ __launch_bounds__(256) void wnorm_kernel(const float* __restrict__ kern,
                                                    const float* __restrict__ style,
                                                    const float* __restrict__ scale,
                                                    _Float16* __restrict__ wT) {
  const int khw = blockIdx.x;   // 0..8
  const int b   = blockIdx.y;
  const int co  = threadIdx.x;
  const float rs = scale[b * COUTC + co];
  _Float16* dst = wT + ((size_t)(b * COUTC + co)) * KTOT + khw * CINC;
  for (int ci = 0; ci < CINC; ++ci) {
    float s = style[b * CINC + ci] + 1.0f;
    float v = kern[((size_t)(khw * CINC + ci)) * COUTC + co] * s * rs;
    dst[ci] = (_Float16)v;
  }
}

// ------------- kernel 4: implicit-GEMM conv via v_wmma_f32_16x16x32_f16 -------------
// block = (row h, sample b); 8 waves: 2 over M (32 px) x 4 over N (64 cout)
// A-tile staging uses GLOBAL_LOAD_ASYNC_TO_LDS_B128 (ASYNCcnt), double buffered.
__global__ __launch_bounds__(256) void modconv_wmma_kernel(
    const _Float16* __restrict__ x16,
    const _Float16* __restrict__ wT,
    float* __restrict__ out) {
  // 66 pixel slots (w_src = -1..64) x 32 channels, double buffered
  __shared__ __align__(16) _Float16 aT[2][66 * 32];

  const int h    = blockIdx.x;
  const int b    = blockIdx.y;
  const int tid  = threadIdx.x;
  const int lane = tid & 31;
  const int wv   = tid >> 5;
  const int waveM = wv & 1;
  const int waveN = wv >> 1;
  const int l15   = lane & 15;
  const int c0    = (lane < 16) ? 0 : 8;   // ISA 16-bit A/B fragment K sub-chunk base

  // padding slots (w_src = -1 and w_src = 64) are always zero; write once for both buffers
  if (tid < 16) {
    const int zbuf  = tid >> 3;
    const int zslot = (tid & 4) ? 65 : 0;
    const int zch   = (tid & 3) * 8;
    v8h z = {};
    *(v8h*)&aT[zbuf][zslot * 32 + zch] = z;
  }

  const int sw = tid >> 2;        // staged source pixel w (0..63) -> slot sw+1
  const int sc = (tid & 3) * 8;   // staged channel sub-offset within the 32-chunk

  // Issue staging for stage si into buffer si&1.
  // Valid row  -> async copy global->LDS (returns 1, tracked by ASYNCcnt)
  // Invalid row-> LDS zero fill (returns 0, ordered by __syncthreads/DScnt)
  // hsrc validity is uniform across the block, so this branch is wave-uniform.
  auto issue_stage = [&](int si) -> int {
    const int kh   = si >> 3;     // 0..2
    const int cc   = si & 7;      // ci chunk 0..7
    const int hsrc = h + kh - 1;
    _Float16* lp = &aT[si & 1][(sw + 1) * 32 + sc];
    if (hsrc >= 0 && hsrc < HH) {
      const _Float16* gp =
          x16 + (((size_t)(b * HH + hsrc) * WW + sw) * CINC) + cc * 32 + sc;
      unsigned laddr = (unsigned)(uintptr_t)lp;              // LDS byte offset
      unsigned long long gaddr = (unsigned long long)(uintptr_t)gp;
      asm volatile("global_load_async_to_lds_b128 %0, %1, off"
                   :
                   : "v"(laddr), "v"(gaddr)
                   : "memory");
      return 1;
    } else {
      v8h z = {};
      *(v8h*)lp = z;
      return 0;
    }
  };

  int pend = issue_stage(0);

  v8f acc[2][4] = {};
  const _Float16* wbase = wT + (size_t)(b * COUTC) * KTOT;

  for (int si = 0; si < 24; ++si) {
    const int buf = si & 1;

    int pendNext = 0;
    if (si < 23) pendNext = issue_stage(si + 1);  // fills buf^1, fenced by last barrier

    // retire stage si (in-order async completion; the 1 outstanding op is stage si+1)
    if (pend) {
      if (pendNext) asm volatile("s_wait_asynccnt 0x1" ::: "memory");
      else          asm volatile("s_wait_asynccnt 0x0" ::: "memory");
    }
    __syncthreads();

    const int kh = si >> 3;
    const int cc = si & 7;
#pragma unroll
    for (int kw = 0; kw < 3; ++kw) {
      const int k0 = (kh * 3 + kw) * CINC + cc * 32;

      // A fragments from LDS (ISA 16-bit A 16x32 layout: chunks at K = c0 and c0+16)
      v16h aF[2];
#pragma unroll
      for (int mt = 0; mt < 2; ++mt) {
        const int slot = waveM * 32 + mt * 16 + l15 + kw;  // w_src + 1
        v8h lo = *(const v8h*)&aT[buf][slot * 32 + c0];
        v8h hi = *(const v8h*)&aT[buf][slot * 32 + c0 + 16];
        aF[mt] = __builtin_shufflevector(lo, hi, 0, 1, 2, 3, 4, 5, 6, 7,
                                         8, 9, 10, 11, 12, 13, 14, 15);
      }
#pragma unroll
      for (int nt = 0; nt < 4; ++nt) {
        const int co = waveN * 64 + nt * 16 + l15;
        const _Float16* bp = wbase + (size_t)co * KTOT + k0 + c0;
        v8h lo = *(const v8h*)bp;
        v8h hi = *(const v8h*)(bp + 16);
        v16h bF = __builtin_shufflevector(lo, hi, 0, 1, 2, 3, 4, 5, 6, 7,
                                          8, 9, 10, 11, 12, 13, 14, 15);
#pragma unroll
        for (int mt = 0; mt < 2; ++mt) {
          acc[mt][nt] = __builtin_amdgcn_wmma_f32_16x16x32_f16(
              false, aF[mt], false, bF, (short)0, acc[mt][nt], false, false);
        }
      }
    }

    __syncthreads();   // all waves done reading buf before next iter overwrites it
    pend = pendNext;
  }

  // D layout: VGPR r -> M = r (lanes 0-15) / r+8 (lanes 16-31); N = lane&15
  const int mOfs = (lane >= 16) ? 8 : 0;
  float* outRow = out + ((size_t)(b * HH + h) * WW) * COUTC;
#pragma unroll
  for (int mt = 0; mt < 2; ++mt) {
#pragma unroll
    for (int nt = 0; nt < 4; ++nt) {
      const int co = waveN * 64 + nt * 16 + l15;
#pragma unroll
      for (int r = 0; r < 8; ++r) {
        const int w = waveM * 32 + mt * 16 + r + mOfs;
        outRow[(size_t)w * COUTC + co] = acc[mt][nt][r];
      }
    }
  }
}

extern "C" void kernel_launch(void* const* d_in, const int* in_sizes, int n_in,
                              void* d_out, int out_size, void* d_ws, size_t ws_size,
                              hipStream_t stream) {
  (void)in_sizes; (void)n_in; (void)out_size; (void)ws_size;

  const float* x     = (const float*)d_in[0];   // [B,H,W,Cin] f32
  const float* style = (const float*)d_in[1];   // [B,Cin] f32
  const float* kern  = (const float*)d_in[2];   // [KH,KW,Cin,Cout] f32
  float* out = (float*)d_out;                   // [B,H,W,Cout] f32

  char* ws = (char*)d_ws;
  const size_t x16_bytes = (size_t)BATCH * HH * WW * CINC * 2;       // 32 MiB
  const size_t wT_bytes  = (size_t)BATCH * COUTC * KTOT * 2;         // 18 MiB
  _Float16* x16 = (_Float16*)ws;
  _Float16* wT  = (_Float16*)(ws + x16_bytes);
  float* scale  = (float*)(ws + x16_bytes + wT_bytes);               // 16 KiB

  const size_t nx = (size_t)BATCH * HH * WW * CINC;                  // 16,777,216
  cvt_x_kernel<<<dim3((unsigned)(nx / (256 * 8))), 256, 0, stream>>>(x, x16);
  sumsq_kernel<<<dim3(BATCH), 256, 0, stream>>>(kern, style, scale);
  wnorm_kernel<<<dim3(9, BATCH), 256, 0, stream>>>(kern, style, scale, wT);
  modconv_wmma_kernel<<<dim3(HH, BATCH), 256, 0, stream>>>(x16, wT, out);
}
